// GraphSAGE_2800318677197
// MI455X (gfx1250) — matrix operand order repaired and checked by reading, not compiled, not run on hardware
//
#include <hip/hip_runtime.h>

// ---------------------------------------------------------------------------
// GraphSAGE 3-layer inference for MI455X (gfx1250, wave32, WMMA bf16).
//   layer: agg = scatter_sum(x[src] -> dst); mean = agg/max(deg,1)
//          h   = BN(ReLU([mean | x] @ [Wl; Wr] + b))
// Pipeline per layer:
//   1) f32 atomic scatter into agg (L2-resident)
//   2) afeat = bf16([mean | x])   (conversion hoisted OUT of the GEMM)
//   3) WMMA GEMM: weight panel staged in LDS (shared by 8 waves/block),
//      v_wmma_f32_16x16x32_bf16, fused bias/ReLU/BN epilogue
// ---------------------------------------------------------------------------

typedef __attribute__((ext_vector_type(16))) __bf16          v16bf;
typedef __attribute__((ext_vector_type(8)))  __bf16          v8bf;
typedef __attribute__((ext_vector_type(8)))  float           v8f;

__device__ __forceinline__ unsigned int f32_to_bf16_rne(float f) {
  unsigned int u = __float_as_uint(f);
  return (u + 0x7FFFu + ((u >> 16) & 1u)) >> 16;   // round to nearest even
}

__device__ __forceinline__ unsigned int pack_bf16x2(float lo, float hi) {
  return f32_to_bf16_rne(lo) | (f32_to_bf16_rne(hi) << 16);
}

__device__ __forceinline__ void atomic_fadd(float* p, float v) {
  __hip_atomic_fetch_add(p, v, __ATOMIC_RELAXED, __HIP_MEMORY_SCOPE_AGENT);
}

// ------------------------------ utility ------------------------------------

__global__ void fill_zero_kernel(float* __restrict__ p, long n) {
  long i = (long)blockIdx.x * blockDim.x + threadIdx.x;
  long stride = (long)gridDim.x * blockDim.x;
  for (; i < n; i += stride) p[i] = 0.0f;
}

__global__ void degree_kernel(const int* __restrict__ dst, float* __restrict__ deg, int E) {
  int i = blockIdx.x * blockDim.x + threadIdx.x;
  int stride = gridDim.x * blockDim.x;
  for (; i < E; i += stride) atomic_fadd(&deg[dst[i]], 1.0f);
}

// BN eval-mode constants: sc = g*rsqrt(v+eps), sh = be - m*sc
__global__ void bnprep_kernel(const float* __restrict__ g, const float* __restrict__ be,
                              const float* __restrict__ m, const float* __restrict__ v,
                              float* __restrict__ sc, float* __restrict__ sh, int n) {
  int i = blockIdx.x * blockDim.x + threadIdx.x;
  if (i >= n) return;
  float s = g[i] * rsqrtf(v[i] + 1e-5f);
  sc[i] = s;
  sh[i] = be[i] - m[i] * s;
}

// ------------------------- edge scatter (sum) ------------------------------

template <int C>
__global__ void scatter_kernel(const float* __restrict__ x, const int* __restrict__ src,
                               const int* __restrict__ dst, float* __restrict__ agg, int E) {
  constexpr int TPE = C / 4;
  constexpr int EPB = 256 / TPE;
  const int t = threadIdx.x;
  const int e = blockIdx.x * EPB + t / TPE;
  if (e >= E) return;
  const int c = (t % TPE) * 4;
  const int s = src[e], d = dst[e];
  const float4 v = *(const float4*)(x + (size_t)s * C + c);
  float* a = agg + (size_t)d * C + c;
  atomic_fadd(a + 0, v.x);
  atomic_fadd(a + 1, v.y);
  atomic_fadd(a + 2, v.z);
  atomic_fadd(a + 3, v.w);
}

// ---------------- A-matrix prep: afeat = bf16([agg/deg | x]) ---------------

template <int DIN>
__global__ void afeat_prep_kernel(const float* __restrict__ agg, const float* __restrict__ deg,
                                  const float* __restrict__ x, unsigned short* __restrict__ af,
                                  int nrows) {
  constexpr int K2 = 2 * DIN;
  constexpr int CH = K2 / 8;                 // 8-channel chunks per node
  const long idx = (long)blockIdx.x * blockDim.x + threadIdx.x;
  if (idx >= (long)nrows * CH) return;
  const int node = (int)(idx / CH);
  const int c8 = (int)(idx % CH) * 8;
  const float* s;
  float scale;
  if (c8 < DIN) {
    s = agg + (size_t)node * DIN + c8;
    scale = 1.0f / fmaxf(deg[node], 1.0f);   // mean part
  } else {
    s = x + (size_t)node * DIN + (c8 - DIN); // identity part
    scale = 1.0f;
  }
  const float4 v0 = *(const float4*)s;
  const float4 v1 = *(const float4*)(s + 4);
  uint4 u;
  u.x = pack_bf16x2(v0.x * scale, v0.y * scale);
  u.y = pack_bf16x2(v0.z * scale, v0.w * scale);
  u.z = pack_bf16x2(v1.x * scale, v1.y * scale);
  u.w = pack_bf16x2(v1.z * scale, v1.w * scale);
  *(uint4*)(af + (size_t)node * K2 + c8) = u;
}

// ------------------- weight concat + transpose to bf16 ---------------------
// wt[n][k] = bf16( k<DIN ? Wl[k][n] : Wr[k-DIN][n] ); contiguous B fragments.

template <int DIN, int DOUT>
__global__ void wtprep_kernel(const float* __restrict__ wl, const float* __restrict__ wr,
                              unsigned short* __restrict__ wt) {
  constexpr int K2 = 2 * DIN;
  const int idx = blockIdx.x * blockDim.x + threadIdx.x;
  if (idx >= K2 * DOUT) return;
  const int n = idx / K2;
  const int k = idx % K2;
  const float v = (k < DIN) ? wl[(size_t)k * DOUT + n] : wr[(size_t)(k - DIN) * DOUT + n];
  wt[(size_t)n * K2 + k] = (unsigned short)f32_to_bf16_rne(v);
}

// -------------------------- fused SAGE GEMM --------------------------------
// One wave per 16-row strip; NT=8 16x16 output tiles (128 cols) per pass.
// Weight panel K-chunked through LDS, shared by the 8 waves of the block.

template <int DIN, int DOUT, int NT>
__global__ void __launch_bounds__(256)
sage_gemm_kernel(const unsigned short* __restrict__ afeat,
                 const unsigned short* __restrict__ wt,
                 const float* __restrict__ bias, const float* __restrict__ sc,
                 const float* __restrict__ sh, float* __restrict__ out, int nrows) {
  constexpr int K2 = 2 * DIN;
  constexpr int COLS = 16 * NT;              // 128 output cols per pass
  constexpr int KC = 128;                    // K-chunk staged in LDS
  constexpr int LDSS = KC + 8;               // padded stride -> bank-conflict-free
  constexpr int NPASS = DOUT / COLS;
  __shared__ __align__(16) unsigned short lwt[COLS * LDSS];   // ~34.8 KB

  const int lane = threadIdx.x & 31;
  const int wavei = threadIdx.x >> 5;
  const int m0 = (blockIdx.x * 8 + wavei) * 16;
  const int lg = lane >> 4;                  // lane group
  const int lr = lane & 15;
  int arow = m0 + lr;
  if (arow > nrows - 1) arow = nrows - 1;    // clamp tail loads only
  const unsigned short* arowp = afeat + (size_t)arow * K2;
  const bool fullwave = (m0 + 16 <= nrows);  // wave-uniform

  for (int pass = 0; pass < NPASS; ++pass) {
    const int nb = pass * COLS;
    v8f acc[NT] = {};

    for (int kc = 0; kc < K2; kc += KC) {
      __syncthreads();                       // lwt readers from previous chunk done
      // cooperative stage: COLS x KC bf16 (each thread: 8 x 16B copies)
#pragma unroll
      for (int it = 0; it < COLS * (KC / 8) / 256; ++it) {
        const int i = it * 256 + threadIdx.x;
        const int col = i / (KC / 8);
        const int k8 = (i % (KC / 8)) * 8;
        *(uint4*)(lwt + col * LDSS + k8) =
            *(const uint4*)(wt + (size_t)(nb + col) * K2 + kc + k8);
      }
      __syncthreads();

      for (int kb = 0; kb < KC; kb += 32) {
        // A fragment (contiguous K, matching B): lanes 0-15 K{+0..15},
        // lanes 16-31 K{+16..31}. Two raw b128 loads, no VALU.
        const int koff = kc + kb + lg * 16;
        const v8bf a0 = *(const v8bf*)(arowp + koff);
        const v8bf a1 = *(const v8bf*)(arowp + koff + 8);
        const v16bf a = __builtin_shufflevector(a0, a1, 0, 1, 2, 3, 4, 5, 6, 7,
                                                8, 9, 10, 11, 12, 13, 14, 15);

        // B fragments from LDS: lanes 0-15 K{kb..+15}, lanes 16-31 K{kb+16..+31}
        const int bk = kb + lg * 16;
#pragma unroll
        for (int t = 0; t < NT; ++t) {
          const v16bf b = *(const v16bf*)(lwt + (t * 16 + lr) * LDSS + bk);
          acc[t] = __builtin_amdgcn_wmma_f32_16x16x32_bf16(
              false, a, false, b, (short)0, acc[t], false, false);
        }
      }
    }

    // Epilogue: bias + ReLU + BN (precomputed scale/shift)
#pragma unroll
    for (int t = 0; t < NT; ++t) {
      const int col = nb + t * 16 + lr;
      const float bcol = bias[col];
      const float scc = sc[col];
      const float shc = sh[col];
      float* orow = out + (size_t)(m0 + lg * 8) * DOUT + col;
      if (fullwave) {
#pragma unroll
        for (int r = 0; r < 8; ++r) {
          orow[(size_t)r * DOUT] = fmaxf(acc[t][r] + bcol, 0.0f) * scc + shc;
        }
      } else {
#pragma unroll
        for (int r = 0; r < 8; ++r) {
          if (m0 + r + lg * 8 < nrows) {
            orow[(size_t)r * DOUT] = fmaxf(acc[t][r] + bcol, 0.0f) * scc + shc;
          }
        }
      }
    }
  }
}

// ------------------------------ launcher -----------------------------------

extern "C" void kernel_launch(void* const* d_in, const int* in_sizes, int n_in,
                              void* d_out, int out_size, void* d_ws, size_t ws_size,
                              hipStream_t stream) {
  const float* x   = (const float*)d_in[0];
  const int*   ei  = (const int*)d_in[1];
  const float *w1l = (const float*)d_in[2],  *w1r = (const float*)d_in[3],
              *b1  = (const float*)d_in[4],  *g1  = (const float*)d_in[5],
              *be1 = (const float*)d_in[6],  *m1  = (const float*)d_in[7],
              *v1  = (const float*)d_in[8];
  const float *w2l = (const float*)d_in[9],  *w2r = (const float*)d_in[10],
              *b2  = (const float*)d_in[11], *g2  = (const float*)d_in[12],
              *be2 = (const float*)d_in[13], *m2  = (const float*)d_in[14],
              *v2  = (const float*)d_in[15];
  const float *w3l = (const float*)d_in[16], *w3r = (const float*)d_in[17],
              *b3  = (const float*)d_in[18], *g3  = (const float*)d_in[19],
              *be3 = (const float*)d_in[20], *m3  = (const float*)d_in[21],
              *v3  = (const float*)d_in[22];
  (void)n_in; (void)out_size; (void)ws_size;

  const int N = in_sizes[0] / 128;
  const int E = in_sizes[1] / 2;
  const int* src = ei;        // edge_index[0]
  const int* dst = ei + E;    // edge_index[1]

  // workspace carve-out (512B aligned blocks)
  char* ws = (char*)d_ws;
  size_t off = 0;
  auto carve = [&](size_t bytes) -> char* {
    char* p = ws + off;
    off += (bytes + 511) & ~(size_t)511;
    return p;
  };
  float* deg          = (float*)carve((size_t)N * 4);
  float* aggb         = (float*)carve((size_t)N * 256 * 4);
  unsigned short* af  = (unsigned short*)carve((size_t)N * 512 * 2);
  float* h1           = (float*)carve((size_t)N * 128 * 4);
  float* h2           = (float*)carve((size_t)N * 256 * 4);
  unsigned short* wt  = (unsigned short*)carve((size_t)512 * 256 * 2);
  float* scb          = (float*)carve((size_t)256 * 4);
  float* shb          = (float*)carve((size_t)256 * 4);
  float* h3           = (float*)d_out;

  const int TB = 256;
  const int gemm_blocks = ((N + 15) / 16 + 7) / 8;   // 8 waves per block

  // degrees (shared by all layers)
  fill_zero_kernel<<<1024, TB, 0, stream>>>(deg, (long)N);
  degree_kernel<<<2048, TB, 0, stream>>>(dst, deg, E);

  // ---------------- layer 1: 128 -> 128 ----------------
  fill_zero_kernel<<<8192, TB, 0, stream>>>(aggb, (long)N * 128);
  scatter_kernel<128><<<(E + 7) / 8, TB, 0, stream>>>(x, src, dst, aggb, E);
  afeat_prep_kernel<128><<<((long)N * 32 + TB - 1) / TB, TB, 0, stream>>>(aggb, deg, x, af, N);
  wtprep_kernel<128, 128><<<(256 * 128 + TB - 1) / TB, TB, 0, stream>>>(w1l, w1r, wt);
  bnprep_kernel<<<1, 128, 0, stream>>>(g1, be1, m1, v1, scb, shb, 128);
  sage_gemm_kernel<128, 128, 8><<<gemm_blocks, TB, 0, stream>>>(af, wt, b1, scb, shb, h1, N);

  // ---------------- layer 2: 128 -> 256 ----------------
  fill_zero_kernel<<<8192, TB, 0, stream>>>(aggb, (long)N * 128);
  scatter_kernel<128><<<(E + 7) / 8, TB, 0, stream>>>(h1, src, dst, aggb, E);
  afeat_prep_kernel<128><<<((long)N * 32 + TB - 1) / TB, TB, 0, stream>>>(aggb, deg, h1, af, N);
  wtprep_kernel<128, 256><<<(256 * 256 + TB - 1) / TB, TB, 0, stream>>>(w2l, w2r, wt);
  bnprep_kernel<<<1, 256, 0, stream>>>(g2, be2, m2, v2, scb, shb, 256);
  sage_gemm_kernel<128, 256, 8><<<gemm_blocks, TB, 0, stream>>>(af, wt, b2, scb, shb, h2, N);

  // ---------------- layer 3: 256 -> 256 ----------------
  fill_zero_kernel<<<8192, TB, 0, stream>>>(aggb, (long)N * 256);
  scatter_kernel<256><<<(E + 3) / 4, TB, 0, stream>>>(h2, src, dst, aggb, E);
  afeat_prep_kernel<256><<<((long)N * 64 + TB - 1) / TB, TB, 0, stream>>>(aggb, deg, h2, af, N);
  wtprep_kernel<256, 256><<<(512 * 256 + TB - 1) / TB, TB, 0, stream>>>(w3l, w3r, wt);
  bnprep_kernel<<<1, 256, 0, stream>>>(g3, be3, m3, v3, scb, shb, 256);
  sage_gemm_kernel<256, 256, 8><<<gemm_blocks, TB, 0, stream>>>(af, wt, b3, scb, shb, h3, N);
}
